// DATSwinTransformerBlock_32375463477462
// MI455X (gfx1250) — matrix operand order repaired and checked by reading, not compiled, not run on hardware
//
#include <hip/hip_runtime.h>
#include <hip/hip_bf16.h>
#include <cmath>

typedef _Float16 v16h __attribute__((ext_vector_type(16)));
typedef float v8f __attribute__((ext_vector_type(8)));

// ---------------- constants ----------------
// B=4096, N=64 tokens (8x8), C=256, 8 heads (hc=32), 4 groups (gc=64),
// Hk=Wk=4 -> n_sample=16, rpe table 15x15 per head.
// 2 windows per workgroup, 512 threads (16 waves): every WMMA B-fragment
// of the 256x256 weights is fetched from L2 once and reused for 8 WMMAs (Wq, Wo) or 2 (Wk, Wv).

// ---------------- weight pre-pack (f32 -> f16, WMMA-B swizzled) ----------------
// B-fragment k order used here (32x16 f16 tile): lane l holds column n=l%16, K rows
// k = 16*(l/16)+i for i=0..15, contiguous. We store, for each weight,
// tiles indexed (nt, kt): pk[((nt*8+kt)*32 + l)*16 + i] =
//   W[(nt*16 + l%16)*256 + kt*32 + 16*(l/16) + i]
__global__ __launch_bounds__(256) void pack_w_kernel(
    const float* __restrict__ wq, const float* __restrict__ wk,
    const float* __restrict__ wv, const float* __restrict__ wo,
    unsigned short* __restrict__ pk_raw) {
  _Float16* pk = reinterpret_cast<_Float16*>(pk_raw);
  int t = blockIdx.x * 256 + threadIdx.x;   // 16384 threads total
  int w = t >> 12;                          // weight select 0..3
  int rem = t & 4095;                       // (nt*8 + kt)*32 + lane
  int l = rem & 31;
  int ktnt = rem >> 5;
  int kt = ktnt & 7;
  int nt = ktnt >> 3;
  const float* W = (w == 0) ? wq : (w == 1) ? wk : (w == 2) ? wv : wo;
  int row = nt * 16 + (l & 15);
  int col0 = kt * 32 + 16 * (l >> 4);
  _Float16* dst = pk + ((size_t)w << 16) + ((size_t)rem << 4);
  const float* src = W + row * 256 + col0;
#pragma unroll
  for (int i = 0; i < 16; ++i) dst[i] = (_Float16)src[i];
}

// ---------------- fragment loaders ----------------
// A-matrix 16x32 f16 k order used here (same as B): lane l (half hb=l/16, row=l%16), half index i:
//   k = 16*hb + i
// => two contiguous 8-half (16B) chunks at kbase+16*hb and kbase+16*hb+8.
__device__ __forceinline__ v16h load_afrag(const _Float16* base, int row,
                                           int kbase, int hb) {
  union { v16h v; uint4 q[2]; } a;
  const _Float16* p = base + row * 256 + kbase + 16 * hb;
  a.q[0] = *(const uint4*)(p);
  a.q[1] = *(const uint4*)(p + 8);
  return a.v;
}

__device__ __forceinline__ v16h load_bfrag_pk(const _Float16* pk, int nt,
                                              int kt, int lane) {
  union { v16h v; uint4 q[2]; } b;
  const _Float16* p = pk + ((((nt << 3) + kt) << 5) + lane) * 16;
  b.q[0] = *(const uint4*)(p);
  b.q[1] = *(const uint4*)(p + 8);
  return b.v;
}

__device__ __forceinline__ v16h load_bfrag_lds(const _Float16* p) {
  union { v16h v; uint4 q[2]; } b;
  b.q[0] = *(const uint4*)(p);
  b.q[1] = *(const uint4*)(p + 8);
  return b.v;
}

struct Corner { int idx; float w; };
__device__ __forceinline__ Corner mk_corner8(int ix, int iy, float w) {
  bool ok = (ix >= 0) && (ix <= 7) && (iy >= 0) && (iy <= 7);
  int cx = min(max(ix, 0), 7), cy = min(max(iy, 0), 7);
  Corner c;
  c.idx = (cy * 8 + cx) * 256;
  c.w = ok ? w : 0.f;
  return c;
}

__device__ __forceinline__ float rpe_corner(const float* rp, int ix, int iy, float w) {
  bool ok = (ix >= 0) && (ix <= 14) && (iy >= 0) && (iy <= 14);
  int cx = min(max(ix, 0), 14), cy = min(max(iy, 0), 14);
  return ok ? rp[cy * 15 + cx] * w : 0.f;
}

#define WMMA_F16(A, Bf, Cacc) \
  __builtin_amdgcn_wmma_f32_16x16x32_f16(false, (A), false, (Bf), (short)0, (Cacc), false, false)

// ---------------- fused DAT block: 2 windows per workgroup ----------------
__global__ __launch_bounds__(512) void dat_block_kernel(
    const float* __restrict__ x,
    const float* __restrict__ bq, const float* __restrict__ bk,
    const float* __restrict__ bv, const float* __restrict__ bo,
    const float* __restrict__ dww, const float* __restrict__ dwb,
    const float* __restrict__ lnw, const float* __restrict__ lnb,
    const float* __restrict__ pww, const float* __restrict__ rpe,
    const unsigned short* __restrict__ pk_raw,
    float* __restrict__ out) {
  extern __shared__ __align__(128) unsigned char smem[];
  // region map (bytes), w = window 0/1:
  //   [0,65536)         sX[w]  (64x256 f16 each) -> aliased later as sAh[w]
  //   [65536,131072)    sQ[w]  (64x256 f16 each) -> aliased later as sOut[w]
  //   [131072,196608)   sAt[w] (8x64x16 f32 each); also offset-net scratch
  //   [196608,212992)   sXs[w] (16x256 f16 each)
  //   [212992,229376)   sKT[w]
  //   [229376,245760)   sVT[w]
  //   [245760,252960)   sRp (8*15*15 f32)
  //   [252960,253984)   sPos[w] (4*16*2 f32 each)
  _Float16* sX0  = (_Float16*)(smem);
  _Float16* sX1  = (_Float16*)(smem + 32768);
  _Float16* sQ0  = (_Float16*)(smem + 65536);
  _Float16* sQ1  = (_Float16*)(smem + 98304);
  float*    sAt0 = (float*)(smem + 131072);
  float*    sAt1 = (float*)(smem + 163840);
  _Float16* sXs0 = (_Float16*)(smem + 196608);
  _Float16* sXs1 = (_Float16*)(smem + 204800);
  _Float16* sKT0 = (_Float16*)(smem + 212992);
  _Float16* sKT1 = (_Float16*)(smem + 221184);
  _Float16* sVT0 = (_Float16*)(smem + 229376);
  _Float16* sVT1 = (_Float16*)(smem + 237568);
  float*    sRp  = (float*)(smem + 245760);
  float*    sPos0= (float*)(smem + 252960);
  float*    sPos1= (float*)(smem + 253472);
  // aliases
  _Float16* sAh0 = sX0;  _Float16* sAh1 = sX1;     // softmaxed attn (K padded)
  _Float16* sOut0= sQ0;  _Float16* sOut1= sQ1;     // attn*V result

  const _Float16* pkQ = (const _Float16*)pk_raw;
  const _Float16* pkK = pkQ + 65536;
  const _Float16* pkV = pkQ + 131072;
  const _Float16* pkO = pkQ + 196608;

  const int b0 = blockIdx.x * 2;      // two windows: b0, b0+1
  const int tid = (int)threadIdx.x;   // 0..511 (16 waves)
  const int wave = tid >> 5;
  const int lane = tid & 31;
  const int hb = lane >> 4;
  const int ln16 = lane & 15;

  // ---- stage 0: load both windows' X (f32 -> f16 LDS) and rpe table ----
  {
#pragma unroll
    for (int i = 0; i < 16; ++i) {
      int f4 = tid + (i << 9);          // 0..8191 float4s (2 windows)
      int win = f4 >> 12;
      int rest = f4 & 4095;
      const float4* xg = (const float4*)(x + (size_t)(b0 + win) * 16384);
      float4 v = xg[rest];
      int n = rest >> 6;
      int c = (rest & 63) << 2;
      _Float16* d = (win ? sX1 : sX0) + n * 256 + c;
      d[0] = (_Float16)v.x; d[1] = (_Float16)v.y;
      d[2] = (_Float16)v.z; d[3] = (_Float16)v.w;
    }
    for (int j = tid; j < 1800; j += 512) sRp[j] = rpe[j];
  }
  __syncthreads();

  // ---- stage 1: Q = X * Wq^T + bq (both windows) ----
  // One N-tile per wave; each B-fragment loaded from L2 once, 8 WMMAs
  // (4 M-tiles x 2 windows) per fragment.
  {
    int nt = wave;
    v8f a0 = {}, a1 = {}, a2 = {}, a3 = {};
    v8f c0 = {}, c1 = {}, c2 = {}, c3 = {};
#pragma unroll
    for (int kt = 0; kt < 8; ++kt) {
      v16h bf = load_bfrag_pk(pkQ, nt, kt, lane);
      v16h f0 = load_afrag(sX0,  0 + ln16, kt * 32, hb);
      v16h f1 = load_afrag(sX0, 16 + ln16, kt * 32, hb);
      v16h f2 = load_afrag(sX0, 32 + ln16, kt * 32, hb);
      v16h f3 = load_afrag(sX0, 48 + ln16, kt * 32, hb);
      a0 = WMMA_F16(f0, bf, a0);
      a1 = WMMA_F16(f1, bf, a1);
      a2 = WMMA_F16(f2, bf, a2);
      a3 = WMMA_F16(f3, bf, a3);
      v16h g0 = load_afrag(sX1,  0 + ln16, kt * 32, hb);
      v16h g1 = load_afrag(sX1, 16 + ln16, kt * 32, hb);
      v16h g2 = load_afrag(sX1, 32 + ln16, kt * 32, hb);
      v16h g3 = load_afrag(sX1, 48 + ln16, kt * 32, hb);
      c0 = WMMA_F16(g0, bf, c0);
      c1 = WMMA_F16(g1, bf, c1);
      c2 = WMMA_F16(g2, bf, c2);
      c3 = WMMA_F16(g3, bf, c3);
    }
    int d = nt * 16 + ln16;
    float bias = bq[d];
#pragma unroll
    for (int r = 0; r < 8; ++r) {
      int m = r + 8 * hb;
      sQ0[(m +  0) * 256 + d] = (_Float16)(a0[r] + bias);
      sQ0[(m + 16) * 256 + d] = (_Float16)(a1[r] + bias);
      sQ0[(m + 32) * 256 + d] = (_Float16)(a2[r] + bias);
      sQ0[(m + 48) * 256 + d] = (_Float16)(a3[r] + bias);
      sQ1[(m +  0) * 256 + d] = (_Float16)(c0[r] + bias);
      sQ1[(m + 16) * 256 + d] = (_Float16)(c1[r] + bias);
      sQ1[(m + 32) * 256 + d] = (_Float16)(c2[r] + bias);
      sQ1[(m + 48) * 256 + d] = (_Float16)(c3[r] + bias);
    }
  }
  __syncthreads();

  // ---- stage 2: offset network -> sPos[w] ----
  if (tid < 128) {
    int win = tid >> 6;
    int t2 = tid & 63;
    int g = t2 >> 4, s = t2 & 15;
    int hk = s >> 2, wk = s & 3;
    const _Float16* q = win ? sQ1 : sQ0;
    float* tmp = (win ? sAt1 : sAt0) + t2 * 64;   // conv scratch
    for (int cc = 0; cc < 64; ++cc) {
      int d = g * 64 + cc;
      float acc = dwb[cc];
#pragma unroll
      for (int dy = 0; dy < 3; ++dy) {
        int hy = 2 * hk - 1 + dy;
        if (hy < 0 || hy > 7) continue;
#pragma unroll
        for (int dx = 0; dx < 3; ++dx) {
          int wx = 2 * wk - 1 + dx;
          if (wx < 0 || wx > 7) continue;
          acc += dww[cc * 9 + dy * 3 + dx] * (float)q[(hy * 8 + wx) * 256 + d];
        }
      }
      tmp[cc] = acc;
    }
    float mu = 0.f;
    for (int cc = 0; cc < 64; ++cc) mu += tmp[cc];
    mu *= (1.f / 64.f);
    float var = 0.f;
    for (int cc = 0; cc < 64; ++cc) { float dv = tmp[cc] - mu; var += dv * dv; }
    var *= (1.f / 64.f);
    float inv = rsqrtf(var + 1e-5f);
    float offy = 0.f, offx = 0.f;
    for (int cc = 0; cc < 64; ++cc) {
      float tn = (tmp[cc] - mu) * inv * lnw[cc] + lnb[cc];
      float ge = 0.5f * tn * (1.f + erff(tn * 0.70710678118654752f));
      offy += ge * pww[cc];
      offx += ge * pww[64 + cc];
    }
    // tanh * (1/(Hk-1)) * OFF_FACTOR = tanh * 2/3 ; ref point (y,x)
    offy = tanhf(offy) * (2.f / 3.f);
    offx = tanhf(offx) * (2.f / 3.f);
    float ry = ((float)hk + 0.5f) * 0.5f - 1.f;
    float rx = ((float)wk + 0.5f) * 0.5f - 1.f;
    float* ps = win ? sPos1 : sPos0;
    ps[t2 * 2 + 0] = offy + ry;
    ps[t2 * 2 + 1] = offx + rx;
  }
  __syncthreads();

  // ---- stage 3: bilinear sample xs^T (16 samples x 256 ch per window) ----
  {
    int win = tid >> 8;
    int t2 = tid & 255;
    int gs = t2 >> 2;              // (g,s) pair, 64 of them
    int cc0 = (t2 & 3) << 4;       // 16-channel slice
    int g = gs >> 4, s = gs & 15;
    const float* ps = win ? sPos1 : sPos0;
    const _Float16* xs = win ? sX1 : sX0;
    _Float16* dst = win ? sXs1 : sXs0;
    float py = ps[gs * 2 + 0], px = ps[gs * 2 + 1];
    float gx = (px + 1.f) * 0.5f * 7.f;
    float gy = (py + 1.f) * 0.5f * 7.f;
    float x0f = floorf(gx), y0f = floorf(gy);
    int x0 = (int)x0f, y0 = (int)y0f;
    float wx1 = gx - x0f, wy1 = gy - y0f;
    Corner c00 = mk_corner8(x0,     y0,     (1.f - wx1) * (1.f - wy1));
    Corner c10 = mk_corner8(x0 + 1, y0,     wx1 * (1.f - wy1));
    Corner c01 = mk_corner8(x0,     y0 + 1, (1.f - wx1) * wy1);
    Corner c11 = mk_corner8(x0 + 1, y0 + 1, wx1 * wy1);
    for (int cc = cc0; cc < cc0 + 16; ++cc) {
      int c = g * 64 + cc;
      float v = c00.w * (float)xs[c00.idx + c] + c10.w * (float)xs[c10.idx + c]
              + c01.w * (float)xs[c01.idx + c] + c11.w * (float)xs[c11.idx + c];
      dst[s * 256 + c] = (_Float16)v;
    }
  }
  __syncthreads();

  // ---- stage 4: K^T, V^T = xs^T * W^T + bias (both windows share B-frag) ----
  for (int job = wave; job < 32; job += 16) {
    int wsel = job >> 4, nt = job & 15;
    const _Float16* pk = wsel ? pkV : pkK;
    const float* bias_v = wsel ? bv : bk;
    _Float16* d0 = wsel ? sVT0 : sKT0;
    _Float16* d1 = wsel ? sVT1 : sKT1;
    v8f acc0 = {}, acc1 = {};
#pragma unroll
    for (int kt = 0; kt < 8; ++kt) {
      v16h bf = load_bfrag_pk(pk, nt, kt, lane);
      v16h f0 = load_afrag(sXs0, ln16, kt * 32, hb);
      v16h f1 = load_afrag(sXs1, ln16, kt * 32, hb);
      acc0 = WMMA_F16(f0, bf, acc0);
      acc1 = WMMA_F16(f1, bf, acc1);
    }
    int d = nt * 16 + ln16;
    float bias = bias_v[d];
#pragma unroll
    for (int r = 0; r < 8; ++r) {
      int s = r + 8 * hb;
      d0[s * 256 + d] = (_Float16)(acc0[r] + bias);
      d1[s * 256 + d] = (_Float16)(acc1[r] + bias);
    }
  }
  __syncthreads();

  // ---- stage 5a: attn = Q_h K_h^T * scale + rpe-bias -> sAt[w] ----
  for (int job = wave; job < 64; job += 16) {
    int win = job >> 5;
    int rest = job & 31;
    int hd = rest >> 2, mt = rest & 3;
    const _Float16* q = win ? sQ1 : sQ0;
    const _Float16* kT = win ? sKT1 : sKT0;
    const float* ps = win ? sPos1 : sPos0;
    float* at = win ? sAt1 : sAt0;
    v8f acc = {};
    v16h a = load_afrag(q, mt * 16 + ln16, hd * 32, hb);
    v16h bf = load_bfrag_lds(kT + ln16 * 256 + hd * 32 + 16 * hb);
    acc = WMMA_F16(a, bf, acc);
    int g = hd >> 1;
    const float* rp = sRp + hd * 225;
    int s = ln16;
    float py = ps[(g * 16 + s) * 2 + 0], px = ps[(g * 16 + s) * 2 + 1];
#pragma unroll
    for (int r = 0; r < 8; ++r) {
      int m = mt * 16 + r + 8 * hb;
      float qy = ((float)(m >> 3) + 0.5f) * 0.25f - 1.f;
      float qx = ((float)(m & 7) + 0.5f) * 0.25f - 1.f;
      float dy = (qy - py) * 0.5f, dx = (qx - px) * 0.5f;
      float gy = (dy + 1.f) * 0.5f * 14.f;
      float gx = (dx + 1.f) * 0.5f * 14.f;
      float x0f = floorf(gx), y0f = floorf(gy);
      int x0 = (int)x0f, y0 = (int)y0f;
      float wx1 = gx - x0f, wy1 = gy - y0f;
      float bias = rpe_corner(rp, x0,     y0,     (1.f - wx1) * (1.f - wy1))
                 + rpe_corner(rp, x0 + 1, y0,     wx1 * (1.f - wy1))
                 + rpe_corner(rp, x0,     y0 + 1, (1.f - wx1) * wy1)
                 + rpe_corner(rp, x0 + 1, y0 + 1, wx1 * wy1);
      at[(hd * 64 + m) * 16 + s] = acc[r] * 0.17677669529663687f + bias;
    }
  }
  __syncthreads();

  // ---- stage 5b: softmax over 16 samples; pack f16 A (K padded to 32) ----
#pragma unroll
  for (int it = 0; it < 2; ++it) {
    int row = tid + (it << 9);        // 1024 rows = 2 win * 8 heads * 64 q
    int win = row >> 9;
    int r2 = row & 511;
    int hd = r2 >> 6, m = r2 & 63;
    const float* src = (win ? sAt1 : sAt0) + r2 * 16;
    float vals[16];
    float mx = -1e30f;
#pragma unroll
    for (int s2 = 0; s2 < 16; ++s2) { vals[s2] = src[s2]; mx = fmaxf(mx, vals[s2]); }
    float sum = 0.f;
#pragma unroll
    for (int s2 = 0; s2 < 16; ++s2) { vals[s2] = __expf(vals[s2] - mx); sum += vals[s2]; }
    float rinv = 1.f / sum;
    _Float16* dst = (win ? sAh1 : sAh0) + m * 256 + hd * 32;  // aliases sX[w]
#pragma unroll
    for (int s2 = 0; s2 < 16; ++s2) {
      dst[s2] = (_Float16)(vals[s2] * rinv);
      dst[16 + s2] = (_Float16)0.f;             // zero-pad K 16..31
    }
  }
  __syncthreads();

  // ---- stage 5c: OUT = attn * V^T (per-head, K=16 padded to 32) ----
  for (int job = wave; job < 128; job += 16) {
    int win = job >> 6;
    int rest = job & 63;
    int mt = rest >> 4, nt = rest & 15;
    int hd = nt >> 1;
    const _Float16* ah = win ? sAh1 : sAh0;
    const _Float16* vT = win ? sVT1 : sVT0;
    _Float16* ot = win ? sOut1 : sOut0;
    v16h a = load_afrag(ah, mt * 16 + ln16, hd * 32, hb);
    union { v16h v; _Float16 e[16]; } bu;
    int d = nt * 16 + ln16;
#pragma unroll
    for (int i = 0; i < 16; ++i) {
      _Float16 vv = vT[i * 256 + d];            // B[k=s, n=d] = VT(s, d)
      bu.e[i] = hb ? (_Float16)0.f : vv;        // rows k=16..31 are zero
    }
    v8f acc = {};
    acc = WMMA_F16(a, bu.v, acc);
#pragma unroll
    for (int r = 0; r < 8; ++r) {
      int m = mt * 16 + r + 8 * hb;
      ot[m * 256 + d] = (_Float16)acc[r];       // aliases sQ[w]
    }
  }
  __syncthreads();

  // ---- stage 6: Y = OUT * Wo^T + bo ; store (B,C,H,W), both windows ----
  {
    int nt = wave;
    v8f a0 = {}, a1 = {}, a2 = {}, a3 = {};
    v8f c0 = {}, c1 = {}, c2 = {}, c3 = {};
#pragma unroll
    for (int kt = 0; kt < 8; ++kt) {
      v16h bf = load_bfrag_pk(pkO, nt, kt, lane);
      v16h f0 = load_afrag(sOut0,  0 + ln16, kt * 32, hb);
      v16h f1 = load_afrag(sOut0, 16 + ln16, kt * 32, hb);
      v16h f2 = load_afrag(sOut0, 32 + ln16, kt * 32, hb);
      v16h f3 = load_afrag(sOut0, 48 + ln16, kt * 32, hb);
      a0 = WMMA_F16(f0, bf, a0);
      a1 = WMMA_F16(f1, bf, a1);
      a2 = WMMA_F16(f2, bf, a2);
      a3 = WMMA_F16(f3, bf, a3);
      v16h g0 = load_afrag(sOut1,  0 + ln16, kt * 32, hb);
      v16h g1 = load_afrag(sOut1, 16 + ln16, kt * 32, hb);
      v16h g2 = load_afrag(sOut1, 32 + ln16, kt * 32, hb);
      v16h g3 = load_afrag(sOut1, 48 + ln16, kt * 32, hb);
      c0 = WMMA_F16(g0, bf, c0);
      c1 = WMMA_F16(g1, bf, c1);
      c2 = WMMA_F16(g2, bf, c2);
      c3 = WMMA_F16(g3, bf, c3);
    }
    int d = nt * 16 + ln16;
    float bias = bo[d];
#pragma unroll
    for (int win = 0; win < 2; ++win) {
      float* dst = out + (size_t)(b0 + win) * 16384 + d * 64 + 8 * hb;
      v8f t0 = win ? c0 : a0;
      v8f t1 = win ? c1 : a1;
      v8f t2 = win ? c2 : a2;
      v8f t3 = win ? c3 : a3;
      {
        float4 lo = { t0[0] + bias, t0[1] + bias, t0[2] + bias, t0[3] + bias };
        float4 hi = { t0[4] + bias, t0[5] + bias, t0[6] + bias, t0[7] + bias };
        *(float4*)(dst +  0) = lo; *(float4*)(dst +  4) = hi;
      }
      {
        float4 lo = { t1[0] + bias, t1[1] + bias, t1[2] + bias, t1[3] + bias };
        float4 hi = { t1[4] + bias, t1[5] + bias, t1[6] + bias, t1[7] + bias };
        *(float4*)(dst + 16) = lo; *(float4*)(dst + 20) = hi;
      }
      {
        float4 lo = { t2[0] + bias, t2[1] + bias, t2[2] + bias, t2[3] + bias };
        float4 hi = { t2[4] + bias, t2[5] + bias, t2[6] + bias, t2[7] + bias };
        *(float4*)(dst + 32) = lo; *(float4*)(dst + 36) = hi;
      }
      {
        float4 lo = { t3[0] + bias, t3[1] + bias, t3[2] + bias, t3[3] + bias };
        float4 hi = { t3[4] + bias, t3[5] + bias, t3[6] + bias, t3[7] + bias };
        *(float4*)(dst + 48) = lo; *(float4*)(dst + 52) = hi;
      }
    }
  }
}

extern "C" void kernel_launch(void* const* d_in, const int* in_sizes, int n_in,
                              void* d_out, int out_size, void* d_ws, size_t ws_size,
                              hipStream_t stream) {
  const float* x   = (const float*)d_in[0];
  const float* wq  = (const float*)d_in[1];
  const float* bq  = (const float*)d_in[2];
  const float* wk  = (const float*)d_in[3];
  const float* bk  = (const float*)d_in[4];
  const float* wv  = (const float*)d_in[5];
  const float* bv  = (const float*)d_in[6];
  const float* wo  = (const float*)d_in[7];
  const float* bo  = (const float*)d_in[8];
  const float* dww = (const float*)d_in[9];
  const float* dwb = (const float*)d_in[10];
  const float* lnw = (const float*)d_in[11];
  const float* lnb = (const float*)d_in[12];
  const float* pww = (const float*)d_in[13];
  const float* rpe = (const float*)d_in[14];
  unsigned short* pk = (unsigned short*)d_ws;   // 4 * 65536 f16 = 512 KB

  pack_w_kernel<<<64, 256, 0, stream>>>(wq, wk, wv, wo, pk);

  const size_t shmem = 253984;   // see region map in kernel
  dat_block_kernel<<<2048, 512, shmem, stream>>>(
      x, bq, bk, bv, bo, dww, dwb, lnw, lnb, pww, rpe,
      (const unsigned short*)pk, (float*)d_out);
}
